// QFBNNet_44813688767075
// MI455X (gfx1250) — compile-verified, matrix operand order checked by reading, not run on hardware
//
#include <hip/hip_runtime.h>
#include <hip/hip_bf16.h>

typedef _Float16 v16h __attribute__((ext_vector_type(16)));
typedef _Float16 v8h  __attribute__((ext_vector_type(8)));
typedef float    v8f  __attribute__((ext_vector_type(8)));

#if __has_builtin(__builtin_amdgcn_tanhf)
#define FAST_TANH(x) __builtin_amdgcn_tanhf(x)
#else
#define FAST_TANH(x) tanhf(x)
#endif

// ---------------------------------------------------------------------------
// Async global->LDS copy (CDNA5 GLOBAL_LOAD_ASYNC_TO_LDS_B128, ASYNCcnt).
// Builtin signature (from hipcc diagnostic): param0 = int4 vector pointer in
// address_space(1) (printed as '__device__'), param1 = int4 vector pointer in
// address_space(3), then two imm ints (offset, cpol).
// LDS operand: ISA aperture rule says a flat pointer into LDS carries the LDS
// byte offset in addr[31:0], so truncating the flat address yields a valid
// as(3) pointer. Guarded: falls back to vectorized through-VGPR copies.
// ---------------------------------------------------------------------------
#if __has_builtin(__builtin_amdgcn_global_load_async_to_lds_b128)
#define HAVE_ASYNC_LDS 1
typedef int v4i_t __attribute__((vector_size(16)));
typedef __attribute__((address_space(1))) v4i_t* as1_v4i_p;
typedef __attribute__((address_space(3))) v4i_t* as3_v4i_p;
__device__ __forceinline__ void async_cp16(void* lds, const void* g) {
    __builtin_amdgcn_global_load_async_to_lds_b128(
        (as1_v4i_p)(unsigned long long)g,
        (as3_v4i_p)(unsigned int)(unsigned long long)lds,
        0, 0);
}
__device__ __forceinline__ void async_wait0() {
#if __has_builtin(__builtin_amdgcn_s_wait_asynccnt)
    __builtin_amdgcn_s_wait_asynccnt(0);
#else
    asm volatile("s_wait_asynccnt 0x0" ::: "memory");
#endif
}
#else
#define HAVE_ASYNC_LDS 0
#endif

// ---------------------------------------------------------------------------
// Kernel 1/2: fused softmax-over-atoms weight effective matrix.
// atoms = [W, 0, tanh(W), sin(W)], p = softmax(logits / 1.0)
// logits layout: (..., K=4) contiguous -> one aligned float4 per element.
// ---------------------------------------------------------------------------
__device__ __forceinline__ float weff_elem(float w, float4 l) {
    float m  = fmaxf(fmaxf(l.x, l.y), fmaxf(l.z, l.w));
    float e0 = __expf(l.x - m);
    float e1 = __expf(l.y - m);
    float e2 = __expf(l.z - m);
    float e3 = __expf(l.w - m);
    float inv = __frcp_rn(e0 + e1 + e2 + e3);
    return (e0 * w + e2 * FAST_TANH(w) + e3 * __sinf(w)) * inv;
}

__global__ __launch_bounds__(256) void atomize_f32(
    const float* __restrict__ W, const float* __restrict__ logits,
    float* __restrict__ out, int N) {
    int i = blockIdx.x * 256 + threadIdx.x;
    if (i >= N) return;
    out[i] = weff_elem(W[i], reinterpret_cast<const float4*>(logits)[i]);
}

__global__ __launch_bounds__(256) void atomize_f16(
    const float* __restrict__ W, const float* __restrict__ logits,
    _Float16* __restrict__ out, int N) {
    int i = blockIdx.x * 256 + threadIdx.x;
    if (i >= N) return;
    out[i] = (_Float16)weff_elem(W[i], reinterpret_cast<const float4*>(logits)[i]);
}

// ---------------------------------------------------------------------------
// Kernel 3: layer 0 -- h1[b,o] = sum_i tanh(x[b,i] * Weff0[o,i]) + b0[o]
// tanh inside the reduction => no WMMA possible; TRANS-op (v_tanh_f32) bound.
// 32x32 output tile per 256-thread block, 4 outputs/thread. Each thread
// stages exactly one 16B chunk of each operand tile per K step (async
// direct-to-LDS when available). Rows padded to 36 floats (144 B): keeps
// 16B alignment and a conflict-free bank stride.
// ---------------------------------------------------------------------------
__global__ __launch_bounds__(256) void layer0_kernel(
    const float* __restrict__ x, const float* __restrict__ Weff0,
    const float* __restrict__ b0, _Float16* __restrict__ h1,
    int D0, int D1) {
    __shared__ __align__(16) float xs[32][36];
    __shared__ __align__(16) float wl[32][36];
    const int t  = threadIdx.x;
    const int tx = t & 15;          // o within tile
    const int ty = t >> 4;          // b within tile
    const int o0 = blockIdx.x * 32;
    const int bb0 = blockIdx.y * 32;

    const int r  = t >> 3;          // staging row 0..31
    const int c4 = (t & 7) * 4;     // staging col 0,4,...,28

    float a00 = 0.f, a01 = 0.f, a10 = 0.f, a11 = 0.f;

    for (int kt = 0; kt < D0; kt += 32) {
#if HAVE_ASYNC_LDS
        async_cp16(&xs[r][c4], &x[(bb0 + r) * D0 + kt + c4]);
        async_cp16(&wl[r][c4], &Weff0[(o0 + r) * D0 + kt + c4]);
        async_wait0();
#else
        *reinterpret_cast<float4*>(&xs[r][c4]) =
            *reinterpret_cast<const float4*>(&x[(bb0 + r) * D0 + kt + c4]);
        *reinterpret_cast<float4*>(&wl[r][c4]) =
            *reinterpret_cast<const float4*>(&Weff0[(o0 + r) * D0 + kt + c4]);
#endif
        __syncthreads();
#pragma unroll 8
        for (int kk = 0; kk < 32; ++kk) {
            float x0 = xs[ty][kk],      x1 = xs[ty + 16][kk];
            float w0 = wl[tx][kk],      w1 = wl[tx + 16][kk];
            a00 += FAST_TANH(x0 * w0);
            a01 += FAST_TANH(x0 * w1);
            a10 += FAST_TANH(x1 * w0);
            a11 += FAST_TANH(x1 * w1);
        }
        __syncthreads();
    }
    h1[(bb0 + ty)      * D1 + o0 + tx]      = (_Float16)(a00 + b0[o0 + tx]);
    h1[(bb0 + ty)      * D1 + o0 + tx + 16] = (_Float16)(a01 + b0[o0 + tx + 16]);
    h1[(bb0 + ty + 16) * D1 + o0 + tx]      = (_Float16)(a10 + b0[o0 + tx]);
    h1[(bb0 + ty + 16) * D1 + o0 + tx + 16] = (_Float16)(a11 + b0[o0 + tx + 16]);
}

// ---------------------------------------------------------------------------
// Kernel 4: layer 1 GEMM via V_WMMA_F32_16X16X32_F16.
// out[b,n] = sum_k h1[b,k] * Weff1[n,k] + b1[n]   (128 x 512 x 1024)
// Block = 128 threads (4 waves); block tile = 16(M) x 64(N); each wave owns
// one 16x16 D tile. K stepped by 64 through LDS (two WMMA issues per step).
//
// Wave32 fragment layouts (ISA 7.12.2, 16-bit A 16x32):
//   lane l: m = l&15, kh = l>>4
//   A elems 0..7  -> K = kh*8 + e        (one 16B ds_load_b128)
//   A elems 8..15 -> K = 16 + kh*8 + e-8 (one 16B ds_load_b128)
// B is staged N-major in LDS (row = N, col = K) so its gather is the same
// contiguous pattern. Rows padded to 72 halves (144 B) -> conflict-free,
// 16B-aligned ds_load_b128 gathers.
// C/D: VGPR r <-> M = 8*kh + r, N = lane&15.
// ---------------------------------------------------------------------------
__global__ __launch_bounds__(128) void gemm_wmma_kernel(
    const _Float16* __restrict__ A,   // h1:   128 x 1024, row-major
    const _Float16* __restrict__ Bm,  // Weff1: 512 x 1024, row-major (N,K)
    const float* __restrict__ bias, float* __restrict__ out,
    int K, int N) {
    __shared__ __align__(16) _Float16 lA[16][72];
    __shared__ __align__(16) _Float16 lB[64][72];

    const int t    = threadIdx.x;
    const int lane = t & 31;
    const int w    = t >> 5;        // wave id 0..3 -> N sub-tile
    const int m16  = lane & 15;
    const int kh   = lane >> 4;
    const int b0i  = blockIdx.y * 16;
    const int n0   = blockIdx.x * 64;

    const int am = t >> 3;          // A staging row 0..15
    const int akq = (t & 7) * 8;    // A staging col (halves)

    v8f c;
#pragma unroll
    for (int r = 0; r < 8; ++r) c[r] = 0.0f;

    for (int kt = 0; kt < K; kt += 64) {
        // Stage A tile (16 x 64 halves, one 16B chunk/thread) and
        // B tile (64 x 64 halves, four 16B chunks/thread, uniform unroll).
#if HAVE_ASYNC_LDS
        async_cp16(&lA[am][akq], &A[(b0i + am) * K + kt + akq]);
#pragma unroll
        for (int q = 0; q < 4; ++q) {
            int rr = t + q * 128;
            int n = rr >> 3, kq = (rr & 7) * 8;
            async_cp16(&lB[n][kq], &Bm[(n0 + n) * K + kt + kq]);
        }
        async_wait0();
#else
        *reinterpret_cast<v8h*>(&lA[am][akq]) =
            *reinterpret_cast<const v8h*>(&A[(b0i + am) * K + kt + akq]);
#pragma unroll
        for (int q = 0; q < 4; ++q) {
            int rr = t + q * 128;
            int n = rr >> 3, kq = (rr & 7) * 8;
            *reinterpret_cast<v8h*>(&lB[n][kq]) =
                *reinterpret_cast<const v8h*>(&Bm[(n0 + n) * K + kt + kq]);
        }
#endif
        __syncthreads();

#pragma unroll
        for (int sub = 0; sub < 64; sub += 32) {
            union { v16h v; v8h h[2]; } af, bf;
            af.h[0] = *reinterpret_cast<const v8h*>(&lA[m16][sub + kh * 8]);
            af.h[1] = *reinterpret_cast<const v8h*>(&lA[m16][sub + 16 + kh * 8]);
            bf.h[0] = *reinterpret_cast<const v8h*>(&lB[w * 16 + m16][sub + kh * 8]);
            bf.h[1] = *reinterpret_cast<const v8h*>(&lB[w * 16 + m16][sub + 16 + kh * 8]);
            // (neg_a, A, neg_b, B, c_mod, C, reuse_a, reuse_b)
            c = __builtin_amdgcn_wmma_f32_16x16x32_f16(
                    false, af.v, false, bf.v, (short)0, c, false, false);
        }
        __syncthreads();
    }

    const int n  = n0 + w * 16 + m16;
    const float bn = bias[n];
#pragma unroll
    for (int r = 0; r < 8; ++r) {
        int m = kh * 8 + r;
        out[(b0i + m) * N + n] = c[r] + bn;
    }
}

// ---------------------------------------------------------------------------
// Launch. Inputs (setup_inputs order): x, W0, b0, logits0, W1, b1, logits1.
// Workspace layout: Weff0 f32 (4 MB) | h1 f16 (256 KB) | Weff1 f16 (1 MB).
// ---------------------------------------------------------------------------
extern "C" void kernel_launch(void* const* d_in, const int* in_sizes, int n_in,
                              void* d_out, int out_size, void* d_ws, size_t ws_size,
                              hipStream_t stream) {
    const float* x   = (const float*)d_in[0];
    const float* W0  = (const float*)d_in[1];
    const float* b0  = (const float*)d_in[2];
    const float* lg0 = (const float*)d_in[3];
    const float* W1  = (const float*)d_in[4];
    const float* b1  = (const float*)d_in[5];
    const float* lg1 = (const float*)d_in[6];
    float* out = (float*)d_out;

    const int B = 128, D0 = 1024, D1 = 1024, D2 = 512;

    char* ws = (char*)d_ws;
    float*    Weff0 = (float*)ws;                                         // D1*D0 f32
    _Float16* h1    = (_Float16*)(ws + (size_t)D1 * D0 * sizeof(float));  // B*D1 f16
    _Float16* Weff1 = (_Float16*)(ws + (size_t)D1 * D0 * sizeof(float)
                                     + (size_t)B * D1 * sizeof(_Float16)); // D2*D1 f16

    const int N0 = D1 * D0;
    const int N1 = D2 * D1;
    atomize_f32<<<(N0 + 255) / 256, 256, 0, stream>>>(W0, lg0, Weff0, N0);
    atomize_f16<<<(N1 + 255) / 256, 256, 0, stream>>>(W1, lg1, Weff1, N1);

    dim3 g0(D1 / 32, B / 32);
    layer0_kernel<<<g0, 256, 0, stream>>>(x, Weff0, b0, h1, D0, D1);

    dim3 g1(D2 / 64, B / 16);
    gemm_wmma_kernel<<<g1, 128, 0, stream>>>(h1, Weff1, b1, out, D1, D2);
}
